// RandLANet__58789512348283
// MI455X (gfx1250) — matrix operand order repaired and checked by reading, not compiled, not run on hardware
//
#include <hip/hip_runtime.h>
#include <hip/hip_bf16.h>
#include <math.h>

typedef __attribute__((ext_vector_type(16))) _Float16 v16h;
typedef __attribute__((ext_vector_type(8)))  _Float16 v8h;
typedef __attribute__((ext_vector_type(8)))  float    v8f;
typedef __attribute__((ext_vector_type(4)))  unsigned int v4u;

#define NPTS   16384
#define BN     65536     // B*N
#define KNB    16
#define DIN    64
#define D2     64
#define DOUT   128
#define DFIN   256

__device__ __forceinline__ float lrelu(float x) { return x > 0.f ? x : 0.2f * x; }

// 16-bit A-fragment (16x32): lane row = l%16, halves {kb..kb+7},{kb+16..kb+23}
__device__ __forceinline__ v16h load_afrag(const _Float16* p) {
  v8h lo = *(const v8h*)(p);
  v8h hi = *(const v8h*)(p + 16);
  return __builtin_shufflevector(lo, hi, 0, 1, 2, 3, 4, 5, 6, 7,
                                 8, 9, 10, 11, 12, 13, 14, 15);
}

// ---------------------------------------------------------------------------
// Kernel 0: pre-swizzle W[krows][ntiles*16] (f32 row-major) into fragment-major
// f16 covering kcs*32 K-rows (zero-padded past krows):
// out[((t*kcs+kc)*32+lane)*16+j] = row<krows ? W[row][col] : 0
// with row = kc*32+8*(lane>>4)+(j&7)+16*(j>>3), col = t*16+(lane&15).
// ---------------------------------------------------------------------------
__global__ void k0_swizzle(const float* __restrict__ W, _Float16* __restrict__ out,
                           int ntiles, int kcs, int ncols, int krows) {
  const int total = ntiles * kcs * 32 * 16;
  for (int i = threadIdx.x + blockIdx.x * blockDim.x; i < total;
       i += gridDim.x * blockDim.x) {
    const int j = i & 15;
    const int lane = (i >> 4) & 31;
    const int rest = i >> 9;
    const int kc = rest % kcs;
    const int t = rest / kcs;
    const int row = kc * 32 + ((lane >> 4) << 3) + (j & 7) + ((j >> 3) << 4);
    const int col = t * 16 + (lane & 15);
    out[i] = (row < krows) ? (_Float16)W[row * ncols + col] : (_Float16)0.f;
  }
}

// ---------------------------------------------------------------------------
// Kernel 1: f_pc = lrelu(features @ W_mlp1 + b_mlp1)   [BN,64] @ [64,64]
// ---------------------------------------------------------------------------
#define K1_WAVES 8
#define K1_LOOP  8
__global__ __launch_bounds__(256) void k1_mlp1(
    const float* __restrict__ feat, const _Float16* __restrict__ Ws,
    const float* __restrict__ bias, float* __restrict__ fpc) {
  __shared__ __align__(16) _Float16 sW[4 * 2 * 32 * 16];   // 8 KB, swizzled
  for (int i = threadIdx.x; i < (4 * 2 * 32 * 16) / 8; i += blockDim.x)
    ((v4u*)sW)[i] = ((const v4u*)Ws)[i];
  __syncthreads();
  const int lane = threadIdx.x & 31, wav = threadIdx.x >> 5;
  const int half = lane >> 4, l16 = lane & 15;
  v16h Bf[4][2];
  #pragma unroll
  for (int t = 0; t < 4; ++t)
    #pragma unroll
    for (int kc = 0; kc < 2; ++kc)
      Bf[t][kc] = *(const v16h*)(sW + ((t * 2 + kc) * 32 + lane) * 16);
  const int gw = blockIdx.x * K1_WAVES + wav;
  for (int it = 0; it < K1_LOOP; ++it) {
    const int m0 = (gw * K1_LOOP + it) * 16;
    const float* fr = feat + (size_t)(m0 + l16) * DIN;
    v16h A[2];
    #pragma unroll
    for (int kc = 0; kc < 2; ++kc) {
      const int kb = kc * 32 + half * 8;
      #pragma unroll
      for (int j = 0; j < 8; ++j) {
        A[kc][j]     = (_Float16)fr[kb + j];
        A[kc][8 + j] = (_Float16)fr[kb + 16 + j];
      }
    }
    #pragma unroll
    for (int t = 0; t < 4; ++t) {
      const int col = t * 16 + l16;
      v8f acc = {};
      #pragma unroll
      for (int kc = 0; kc < 2; ++kc)
        acc = __builtin_amdgcn_wmma_f32_16x16x32_f16(false, A[kc], false, Bf[t][kc],
                                                     (short)0, acc, false, false);
      const float bc = bias[col];
      #pragma unroll
      for (int r = 0; r < 8; ++r)
        fpc[(size_t)(m0 + r + half * 8) * D2 + col] = lrelu(acc[r] + bc);
    }
  }
}

// ---------------------------------------------------------------------------
// Kernel 2: LocSE stage 1 + attentive pooling 1. One wave per point;
// position encoding and the agg @ W_agg1 tail are WMMA too.
// ---------------------------------------------------------------------------
#define K2_WAVES 4
#define K2_PPW   8
__global__ __launch_bounds__(128) void k2_stage1(
    const float* __restrict__ xyz, const int* __restrict__ nidx,
    const float* __restrict__ fpc,
    const _Float16* __restrict__ Wencs, const float* __restrict__ benc,
    const _Float16* __restrict__ Watts, const float* __restrict__ batt,
    const _Float16* __restrict__ Wagg1s, const float* __restrict__ bagg,
    float* __restrict__ fagg_out) {
  __shared__ __align__(16) _Float16 sW[8 * 4 * 32 * 16];      // 32 KB, swizzled
  __shared__ __align__(16) _Float16 sFC[K2_WAVES][16 * 128];
  __shared__ __align__(16) _Float16 sAgg[K2_WAVES][8 * 128];  // agg rows (f16)
  for (int i = threadIdx.x; i < (8 * 4 * 32 * 16) / 8; i += blockDim.x)
    ((v4u*)sW)[i] = ((const v4u*)Watts)[i];
  __syncthreads();
  const int lane = threadIdx.x & 31, wav = threadIdx.x >> 5;
  const int half = lane >> 4, l16 = lane & 15;
  _Float16* fc = sFC[wav];
  _Float16* aggB = sAgg[wav];
  // hoisted: position-encoding B-fragments (K=10 padded to 32) + biases
  v16h Benc[4];
  float bencr[4], battr[8];
  #pragma unroll
  for (int t = 0; t < 4; ++t) {
    Benc[t] = *(const v16h*)(Wencs + (t * 32 + lane) * 16);
    bencr[t] = benc[t * 16 + l16];
  }
  #pragma unroll
  for (int t = 0; t < 8; ++t) battr[t] = batt[t * 16 + l16];
  const int gw = blockIdx.x * K2_WAVES + wav;
  const int p0 = gw * K2_PPW;
  for (int it = 0; it < K2_PPW; ++it) {
    const int g = p0 + it;
    const int base = (g >> 14) << 14;             // b*N
    // ---- relative position encoding via WMMA (lane k = l16)
    const int k = l16;
    const int ni = nidx[(size_t)g * KNB + k];
    const float* cx = xyz + (size_t)g * 3;
    const float* nx = xyz + (size_t)(base + ni) * 3;
    const float c0 = cx[0], c1 = cx[1], c2 = cx[2];
    const float n0 = nx[0], n1 = nx[1], n2 = nx[2];
    const float rx = c0 - n0, ry = c1 - n1, rz = c2 - n2;
    const float raw[10] = { sqrtf(rx*rx + ry*ry + rz*rz + 1e-12f),
                            rx, ry, rz, c0, c1, c2, n0, n1, n2 };
    v16h Ae = {};
    #pragma unroll
    for (int j = 0; j < 8; ++j) {
      const float v = (half == 0) ? raw[j] : ((j < 2) ? raw[8 + j] : 0.f);
      Ae[j] = (_Float16)v;
    }
    #pragma unroll
    for (int t = 0; t < 4; ++t) {
      v8f acc = {};
      acc = __builtin_amdgcn_wmma_f32_16x16x32_f16(false, Ae, false, Benc[t],
                                                   (short)0, acc, false, false);
      #pragma unroll
      for (int r = 0; r < 8; ++r)
        fc[(r + half * 8) * 128 + 64 + t * 16 + l16] =
            (_Float16)lrelu(acc[r] + bencr[t]);   // f_xyz ch 64..127
    }
    // ---- gather f_pc of neighbor k (ch 0..63)
    const float* fn = fpc + (size_t)(base + ni) * D2;
    #pragma unroll
    for (int cc = 0; cc < 32; cc += 8) {
      v8h vn;
      #pragma unroll
      for (int u = 0; u < 8; ++u) vn[u] = (_Float16)fn[half * 32 + cc + u];
      *(v8h*)(fc + k * 128 + half * 32 + cc) = vn;
    }
    // ---- attention: softmax over 16 neighbors per channel, then agg
    v16h A[4];
    #pragma unroll
    for (int kc = 0; kc < 4; ++kc)
      A[kc] = load_afrag(fc + l16 * 128 + kc * 32 + half * 8);
    #pragma unroll
    for (int t = 0; t < 8; ++t) {
      const int col = t * 16 + l16;
      v8f acc = {};
      #pragma unroll
      for (int kc = 0; kc < 4; ++kc) {
        v16h Bf = *(const v16h*)(sW + ((t * 4 + kc) * 32 + lane) * 16);
        acc = __builtin_amdgcn_wmma_f32_16x16x32_f16(false, A[kc], false, Bf,
                                                     (short)0, acc, false, false);
      }
      const float bc = battr[t];
      float s[8], mx = -3.4e38f;
      #pragma unroll
      for (int r = 0; r < 8; ++r) { s[r] = acc[r] + bc; mx = fmaxf(mx, s[r]); }
      mx = fmaxf(mx, __shfl_xor(mx, 16));
      float e[8], sum = 0.f;
      #pragma unroll
      for (int r = 0; r < 8; ++r) { e[r] = __expf(s[r] - mx); sum += e[r]; }
      sum += __shfl_xor(sum, 16);
      const float inv = 1.f / sum;
      float part = 0.f;
      #pragma unroll
      for (int r = 0; r < 8; ++r)
        part += (float)fc[(r + half * 8) * 128 + col] * (e[r] * inv);
      part += __shfl_xor(part, 16);
      if (half == 0) aggB[it * 128 + col] = (_Float16)part;
    }
  }
  // ---- batched tail: f_pc_agg = lrelu(agg @ W_agg1 + b)  [8(pad16),128]@[128,64]
  {
    v16h A[4];
    #pragma unroll
    for (int kc = 0; kc < 4; ++kc)
      A[kc] = load_afrag(aggB + (l16 & 7) * 128 + kc * 32 + half * 8);
    #pragma unroll
    for (int t = 0; t < 4; ++t) {
      const int col = t * 16 + l16;
      v8f acc = {};
      #pragma unroll
      for (int kc = 0; kc < 4; ++kc) {
        v16h Bf = *(const v16h*)(Wagg1s + ((t * 4 + kc) * 32 + lane) * 16);
        acc = __builtin_amdgcn_wmma_f32_16x16x32_f16(false, A[kc], false, Bf,
                                                     (short)0, acc, false, false);
      }
      const float bc = bagg[col];
      if (half == 0) {
        #pragma unroll
        for (int r = 0; r < 8; ++r)
          fagg_out[(size_t)(p0 + r) * D2 + col] = lrelu(acc[r] + bc);
      }
    }
  }
}

// ---------------------------------------------------------------------------
// Kernel 3: LocSE stage 2 + attentive pooling 2 + residual output.
// ---------------------------------------------------------------------------
#define K3_WAVES 2
#define K3_PPW   8
__global__ __launch_bounds__(64) void k3_stage2(
    const float* __restrict__ xyz, const float* __restrict__ feat,
    const int* __restrict__ nidx, const float* __restrict__ fagg,
    const _Float16* __restrict__ Wencs, const float* __restrict__ benc,
    const _Float16* __restrict__ Wx2s, const float* __restrict__ bxyz2,
    const _Float16* __restrict__ Watt2s, const float* __restrict__ batt2,
    const _Float16* __restrict__ Wagg2s, const float* __restrict__ bagg2,
    const _Float16* __restrict__ Wmlp2s, const float* __restrict__ bmlp2,
    const _Float16* __restrict__ Wshorts, const float* __restrict__ bshort,
    float* __restrict__ out) {
  __shared__ __align__(16) _Float16 sW[8 * 4 * 32 * 16];      // W_att2 swizzled
  __shared__ __align__(16) _Float16 sFX[K3_WAVES][16 * 64];   // f_xyz / lfa buf
  __shared__ __align__(16) _Float16 sFC[K3_WAVES][16 * 128];
  __shared__ __align__(16) _Float16 sAgg[K3_WAVES][8 * 128];
  for (int i = threadIdx.x; i < (8 * 4 * 32 * 16) / 8; i += blockDim.x)
    ((v4u*)sW)[i] = ((const v4u*)Watt2s)[i];
  __syncthreads();
  const int lane = threadIdx.x & 31, wav = threadIdx.x >> 5;
  const int half = lane >> 4, l16 = lane & 15;
  _Float16* fx = sFX[wav];
  _Float16* fc = sFC[wav];
  _Float16* aggB = sAgg[wav];
  v16h Benc[4];
  float bencr[4], battr[8], bx2r[4];
  #pragma unroll
  for (int t = 0; t < 4; ++t) {
    Benc[t] = *(const v16h*)(Wencs + (t * 32 + lane) * 16);
    bencr[t] = benc[t * 16 + l16];
    bx2r[t] = bxyz2[t * 16 + l16];
  }
  #pragma unroll
  for (int t = 0; t < 8; ++t) battr[t] = batt2[t * 16 + l16];
  const int gw = blockIdx.x * K3_WAVES + wav;
  const int p0 = gw * K3_PPW;
  for (int it = 0; it < K3_PPW; ++it) {
    const int g = p0 + it;
    const int base = (g >> 14) << 14;
    // ---- recompute f_xyz via WMMA
    const int k = l16;
    const int ni = nidx[(size_t)g * KNB + k];
    const float* cx = xyz + (size_t)g * 3;
    const float* nx = xyz + (size_t)(base + ni) * 3;
    const float c0 = cx[0], c1 = cx[1], c2 = cx[2];
    const float n0 = nx[0], n1 = nx[1], n2 = nx[2];
    const float rx = c0 - n0, ry = c1 - n1, rz = c2 - n2;
    const float raw[10] = { sqrtf(rx*rx + ry*ry + rz*rz + 1e-12f),
                            rx, ry, rz, c0, c1, c2, n0, n1, n2 };
    v16h Ae = {};
    #pragma unroll
    for (int j = 0; j < 8; ++j) {
      const float v = (half == 0) ? raw[j] : ((j < 2) ? raw[8 + j] : 0.f);
      Ae[j] = (_Float16)v;
    }
    #pragma unroll
    for (int t = 0; t < 4; ++t) {
      v8f acc = {};
      acc = __builtin_amdgcn_wmma_f32_16x16x32_f16(false, Ae, false, Benc[t],
                                                   (short)0, acc, false, false);
      #pragma unroll
      for (int r = 0; r < 8; ++r)
        fx[(r + half * 8) * 64 + t * 16 + l16] = (_Float16)lrelu(acc[r] + bencr[t]);
    }
    // ---- gather f_pc_agg of neighbor k (ch 0..63)
    const float* fn = fagg + (size_t)(base + ni) * D2;
    #pragma unroll
    for (int cc = 0; cc < 32; cc += 8) {
      v8h vn;
      #pragma unroll
      for (int u = 0; u < 8; ++u) vn[u] = (_Float16)fn[half * 32 + cc + u];
      *(v8h*)(fc + k * 128 + half * 32 + cc) = vn;
    }
    // ---- f_xyz2 = lrelu(f_xyz @ W_xyz2 + b)  [16,64]@[64,64]
    {
      v16h A[2];
      #pragma unroll
      for (int kc = 0; kc < 2; ++kc)
        A[kc] = load_afrag(fx + l16 * 64 + kc * 32 + half * 8);
      #pragma unroll
      for (int t = 0; t < 4; ++t) {
        const int col = t * 16 + l16;
        v8f acc = {};
        #pragma unroll
        for (int kc = 0; kc < 2; ++kc) {
          v16h Bf = *(const v16h*)(Wx2s + ((t * 2 + kc) * 32 + lane) * 16);
          acc = __builtin_amdgcn_wmma_f32_16x16x32_f16(false, A[kc], false, Bf,
                                                       (short)0, acc, false, false);
        }
        #pragma unroll
        for (int r = 0; r < 8; ++r)
          fc[(r + half * 8) * 128 + 64 + col] = (_Float16)lrelu(acc[r] + bx2r[t]);
      }
    }
    // ---- attention 2
    v16h A[4];
    #pragma unroll
    for (int kc = 0; kc < 4; ++kc)
      A[kc] = load_afrag(fc + l16 * 128 + kc * 32 + half * 8);
    #pragma unroll
    for (int t = 0; t < 8; ++t) {
      const int col = t * 16 + l16;
      v8f acc = {};
      #pragma unroll
      for (int kc = 0; kc < 4; ++kc) {
        v16h Bf = *(const v16h*)(sW + ((t * 4 + kc) * 32 + lane) * 16);
        acc = __builtin_amdgcn_wmma_f32_16x16x32_f16(false, A[kc], false, Bf,
                                                     (short)0, acc, false, false);
      }
      const float bc = battr[t];
      float s[8], mx = -3.4e38f;
      #pragma unroll
      for (int r = 0; r < 8; ++r) { s[r] = acc[r] + bc; mx = fmaxf(mx, s[r]); }
      mx = fmaxf(mx, __shfl_xor(mx, 16));
      float e[8], sum = 0.f;
      #pragma unroll
      for (int r = 0; r < 8; ++r) { e[r] = __expf(s[r] - mx); sum += e[r]; }
      sum += __shfl_xor(sum, 16);
      const float inv = 1.f / sum;
      float part = 0.f;
      #pragma unroll
      for (int r = 0; r < 8; ++r)
        part += (float)fc[(r + half * 8) * 128 + col] * (e[r] * inv);
      part += __shfl_xor(part, 16);
      if (half == 0) aggB[it * 128 + col] = (_Float16)part;
    }
  }
  // ---- batched tail 1: f_lfa = lrelu(agg @ W_agg2 + b)  [8(pad16),128]@[128,128]
  _Float16* lfaB = fx;                         // reuse dead f_xyz buffer (1024 halfs)
  {
    v16h A[4];
    #pragma unroll
    for (int kc = 0; kc < 4; ++kc)
      A[kc] = load_afrag(aggB + (l16 & 7) * 128 + kc * 32 + half * 8);
    #pragma unroll
    for (int t = 0; t < 8; ++t) {
      const int col = t * 16 + l16;
      v8f acc = {};
      #pragma unroll
      for (int kc = 0; kc < 4; ++kc) {
        v16h Bf = *(const v16h*)(Wagg2s + ((t * 4 + kc) * 32 + lane) * 16);
        acc = __builtin_amdgcn_wmma_f32_16x16x32_f16(false, A[kc], false, Bf,
                                                     (short)0, acc, false, false);
      }
      const float bc = bagg2[col];
      if (half == 0) {
        #pragma unroll
        for (int r = 0; r < 8; ++r)
          lfaB[r * 128 + col] = (_Float16)lrelu(acc[r] + bc);
      }
    }
  }
  // ---- batched tail 2: out = lrelu(lfa @ W_mlp2 + feat @ W_short + biases)
  {
    v16h Al[4], Af[2];
    #pragma unroll
    for (int kc = 0; kc < 4; ++kc)
      Al[kc] = load_afrag(lfaB + (l16 & 7) * 128 + kc * 32 + half * 8);
    const float* fr = feat + (size_t)(p0 + (l16 & 7)) * DIN;
    #pragma unroll
    for (int kc = 0; kc < 2; ++kc) {
      const int kb = kc * 32 + half * 8;
      #pragma unroll
      for (int j = 0; j < 8; ++j) {
        Af[kc][j]     = (_Float16)fr[kb + j];
        Af[kc][8 + j] = (_Float16)fr[kb + 16 + j];
      }
    }
    #pragma unroll
    for (int t = 0; t < 16; ++t) {
      const int col = t * 16 + l16;
      v8f acc = {};
      #pragma unroll
      for (int kc = 0; kc < 4; ++kc) {
        v16h Bf = *(const v16h*)(Wmlp2s + ((t * 4 + kc) * 32 + lane) * 16);
        acc = __builtin_amdgcn_wmma_f32_16x16x32_f16(false, Al[kc], false, Bf,
                                                     (short)0, acc, false, false);
      }
      #pragma unroll
      for (int kc = 0; kc < 2; ++kc) {
        v16h Bf = *(const v16h*)(Wshorts + ((t * 2 + kc) * 32 + lane) * 16);
        acc = __builtin_amdgcn_wmma_f32_16x16x32_f16(false, Af[kc], false, Bf,
                                                     (short)0, acc, false, false);
      }
      const float bc = bmlp2[col] + bshort[col];
      if (half == 0) {
        #pragma unroll
        for (int r = 0; r < 8; ++r)
          out[(size_t)(p0 + r) * DFIN + col] = lrelu(acc[r] + bc);
      }
    }
  }
}

// ---------------------------------------------------------------------------
extern "C" void kernel_launch(void* const* d_in, const int* in_sizes, int n_in,
                              void* d_out, int out_size, void* d_ws, size_t ws_size,
                              hipStream_t stream) {
  (void)in_sizes; (void)n_in; (void)out_size; (void)ws_size;
  const float* xyz     = (const float*)d_in[0];
  const float* feat    = (const float*)d_in[1];
  const int*   nidx    = (const int*)d_in[2];
  const float* W_mlp1  = (const float*)d_in[3];
  const float* b_mlp1  = (const float*)d_in[4];
  const float* W_enc   = (const float*)d_in[5];
  const float* b_enc   = (const float*)d_in[6];
  const float* W_att1  = (const float*)d_in[7];
  const float* b_att1  = (const float*)d_in[8];
  const float* W_agg1  = (const float*)d_in[9];
  const float* b_agg1  = (const float*)d_in[10];
  const float* W_xyz2  = (const float*)d_in[11];
  const float* b_xyz2  = (const float*)d_in[12];
  const float* W_att2  = (const float*)d_in[13];
  const float* b_att2  = (const float*)d_in[14];
  const float* W_agg2  = (const float*)d_in[15];
  const float* b_agg2  = (const float*)d_in[16];
  const float* W_mlp2  = (const float*)d_in[17];
  const float* b_mlp2  = (const float*)d_in[18];
  const float* W_short = (const float*)d_in[19];
  const float* b_short = (const float*)d_in[20];
  float* outp = (float*)d_out;

  float* fpc  = (float*)d_ws;                        // [BN, 64] f32
  float* fagg = fpc + (size_t)BN * D2;               // [BN, 64] f32
  _Float16* Wm1s  = (_Float16*)(fagg + (size_t)BN * D2); // swizzled f16 weights
  _Float16* Wa1s  = Wm1s  + 4 * 2 * 32 * 16;   //  4096 halfs
  _Float16* Wa2s  = Wa1s  + 8 * 4 * 32 * 16;   // 16384
  _Float16* Wx2s  = Wa2s  + 8 * 4 * 32 * 16;   // 16384
  _Float16* Wg1s  = Wx2s  + 4 * 2 * 32 * 16;   //  4096
  _Float16* Wg2s  = Wg1s  + 4 * 4 * 32 * 16;   //  8192
  _Float16* Wm2s  = Wg2s  + 8 * 4 * 32 * 16;   // 16384
  _Float16* Wshs  = Wm2s  + 16 * 4 * 32 * 16;  // 32768
  _Float16* Wens  = Wshs  + 16 * 2 * 32 * 16;  // 16384

  // ---- pre-swizzle all WMMA B-operands to fragment-major f16 (zero-padded K)
  k0_swizzle<<<16, 256, 0, stream>>>(W_mlp1,  Wm1s, 4, 2, D2, 64);
  k0_swizzle<<<32, 256, 0, stream>>>(W_att1,  Wa1s, 8, 4, DOUT, 128);
  k0_swizzle<<<32, 256, 0, stream>>>(W_att2,  Wa2s, 8, 4, DOUT, 128);
  k0_swizzle<<<16, 256, 0, stream>>>(W_xyz2,  Wx2s, 4, 2, D2, 64);
  k0_swizzle<<<16, 256, 0, stream>>>(W_agg1,  Wg1s, 4, 4, D2, 128);
  k0_swizzle<<<32, 256, 0, stream>>>(W_agg2,  Wg2s, 8, 4, DOUT, 128);
  k0_swizzle<<<64, 256, 0, stream>>>(W_mlp2,  Wm2s, 16, 4, DFIN, 128);
  k0_swizzle<<<32, 256, 0, stream>>>(W_short, Wshs, 16, 2, DFIN, 64);
  k0_swizzle<<<8,  256, 0, stream>>>(W_enc,   Wens, 4, 1, D2, 10);

  // k1: 512 waves total, 8 waves/block
  k1_mlp1<<<BN / (16 * K1_LOOP * K1_WAVES), 256, 0, stream>>>(feat, Wm1s, b_mlp1, fpc);
  // k2: one wave per point, 8 points/wave, 4 waves/block
  k2_stage1<<<BN / (K2_PPW * K2_WAVES), 128, 0, stream>>>(
      xyz, nidx, fpc, Wens, b_enc, Wa1s, b_att1, Wg1s, b_agg1, fagg);
  // k3: 2 waves/block (48 KB LDS)
  k3_stage2<<<BN / (K3_PPW * K3_WAVES), 64, 0, stream>>>(
      xyz, feat, nidx, fagg, Wens, b_enc, Wx2s, b_xyz2, Wa2s, b_att2,
      Wg2s, b_agg2, Wm2s, b_mlp2, Wshs, b_short, outp);
}
